// ModifiedHNHNLayer_35845797052899
// MI455X (gfx1250) — matrix operand order repaired and checked by reading, not compiled, not run on hardware
//
#include <hip/hip_runtime.h>

// Problem constants (match the reference)
#define NN 8192      // nodes
#define EE 16384     // hyperedges
#define CC 64        // IN_C == HID
#define PASS_ROWS 32 // rows per block in the stats passes
#define PASS_BLOCKS (NN / PASS_ROWS)   // 256
#define GEMM_ROWS 64 // output rows per block in the WMMA GEMMs

typedef float v8f    __attribute__((ext_vector_type(8)));
typedef __bf16 v16bf __attribute__((ext_vector_type(16)));
typedef short s8v    __attribute__((ext_vector_type(8)));
typedef short s16v   __attribute__((ext_vector_type(16)));

__device__ __forceinline__ short f32_to_bf16_rne(float f) {
    unsigned u = __builtin_bit_cast(unsigned, f);
    u += 0x7fffu + ((u >> 16) & 1u);
    return (short)(u >> 16);
}
// Pack two fp32 -> two bf16 in one dword by truncation (exact for 0.0/1.0).
__device__ __forceinline__ unsigned pack_bf16_trunc(float a, float b) {
    return (__builtin_bit_cast(unsigned, a) >> 16) |
           (__builtin_bit_cast(unsigned, b) & 0xffff0000u);
}

// ---------------------------------------------------------------------------
// Pass 1: one streaming read of B1, float4-vectorized. Per-block LDS column
// accumulator (64KB, atomic-free: thread t owns float4-columns t mod 256),
// wave32 shuffle reduce for row sums. Writes node_norm = node_sum^-0.5 and
// per-block column partials (deterministically reduced later).
// ---------------------------------------------------------------------------
__global__ __launch_bounds__(256) void pass1_kernel(const float* __restrict__ B1,
                                                    float* __restrict__ node_norm,
                                                    float* __restrict__ part) {
    extern __shared__ float smem[];            // EE colacc + 8 wred
    float* colacc = smem;
    float* wred   = smem + EE;
    const int tid = threadIdx.x, w = tid >> 5, lane = tid & 31;
    for (int v = tid; v < EE / 4; v += 256) *(float4*)&colacc[4 * v] = float4{0, 0, 0, 0};
    __syncthreads();
    const int r0 = blockIdx.x * PASS_ROWS;
    for (int r = 0; r < PASS_ROWS; ++r) {
        const float* row = B1 + (size_t)(r0 + r) * EE;
        float rsum = 0.f;
        for (int v = tid; v < EE / 4; v += 256) {
            const float4 f = *(const float4*)(row + 4 * v);
            float4 ca = *(float4*)&colacc[4 * v];
            ca.x += f.x; ca.y += f.y; ca.z += f.z; ca.w += f.w;
            *(float4*)&colacc[4 * v] = ca;
            rsum += (f.x + f.y) + (f.z + f.w);
        }
        for (int off = 16; off > 0; off >>= 1) rsum += __shfl_down(rsum, off, 32);
        if (lane == 0) wred[w] = rsum;
        __syncthreads();
        if (tid == 0) {
            float tot = 0.f;
            for (int ww = 0; ww < 8; ++ww) tot += wred[ww];
            node_norm[r0 + r] = powf(tot, -0.5f);   // BETA = -0.5
        }
        __syncthreads();
    }
    float* pb = part + (size_t)blockIdx.x * EE;
    for (int v = tid; v < EE / 4; v += 256)
        *(float4*)&pb[4 * v] = *(const float4*)&colacc[4 * v];
}

__global__ __launch_bounds__(256) void reduce_edge_norm_kernel(const float* __restrict__ part,
                                                               float* __restrict__ edge_norm) {
    const int j4 = blockIdx.x * 256 + threadIdx.x;     // float4-column index
    float4 s{0, 0, 0, 0};
    for (int b = 0; b < PASS_BLOCKS; ++b) {
        const float4 p = *(const float4*)&part[(size_t)b * EE + 4 * j4];
        s.x += p.x; s.y += p.y; s.z += p.z; s.w += p.w;
    }
    float4 o;
    o.x = powf(s.x, -1.5f); o.y = powf(s.y, -1.5f);    // ALPHA = -1.5
    o.z = powf(s.z, -1.5f); o.w = powf(s.w, -1.5f);
    *(float4*)&edge_norm[4 * j4] = o;
}

// ---------------------------------------------------------------------------
// Pass 2: second streaming read of B1 (mask == B1 since binary).
// rowden[i] = sum_j B1[i,j]*edge_norm[j]  -> row_scale = node_norm/rowden
// colden[j] = sum_i B1[i,j]*node_norm[i]  -> col_scale = edge_norm/colden
// ---------------------------------------------------------------------------
__global__ __launch_bounds__(256) void pass2_kernel(const float* __restrict__ B1,
                                                    const float* __restrict__ edge_norm,
                                                    const float* __restrict__ node_norm,
                                                    float* __restrict__ row_scale,
                                                    float* __restrict__ part) {
    extern __shared__ float smem[];
    float* colacc = smem;
    float* wred   = smem + EE;
    const int tid = threadIdx.x, w = tid >> 5, lane = tid & 31;
    for (int v = tid; v < EE / 4; v += 256) *(float4*)&colacc[4 * v] = float4{0, 0, 0, 0};
    __syncthreads();
    const int r0 = blockIdx.x * PASS_ROWS;
    for (int r = 0; r < PASS_ROWS; ++r) {
        const float* row = B1 + (size_t)(r0 + r) * EE;
        const float nn = node_norm[r0 + r];
        float rsum = 0.f;
        for (int v = tid; v < EE / 4; v += 256) {
            const float4 f  = *(const float4*)(row + 4 * v);
            const float4 en = *(const float4*)&edge_norm[4 * v];  // 64KB, cache-resident
            rsum += (f.x * en.x + f.y * en.y) + (f.z * en.z + f.w * en.w);
            float4 ca = *(float4*)&colacc[4 * v];
            ca.x += f.x * nn; ca.y += f.y * nn; ca.z += f.z * nn; ca.w += f.w * nn;
            *(float4*)&colacc[4 * v] = ca;
        }
        for (int off = 16; off > 0; off >>= 1) rsum += __shfl_down(rsum, off, 32);
        if (lane == 0) wred[w] = rsum;
        __syncthreads();
        if (tid == 0) {
            float tot = 0.f;
            for (int ww = 0; ww < 8; ++ww) tot += wred[ww];
            row_scale[r0 + r] = nn / tot;
        }
        __syncthreads();
    }
    float* pb = part + (size_t)blockIdx.x * EE;
    for (int v = tid; v < EE / 4; v += 256)
        *(float4*)&pb[4 * v] = *(const float4*)&colacc[4 * v];
}

__global__ __launch_bounds__(256) void reduce_col_scale_kernel(const float* __restrict__ part,
                                                               const float* __restrict__ edge_norm,
                                                               float* __restrict__ col_scale) {
    const int j4 = blockIdx.x * 256 + threadIdx.x;
    float4 s{0, 0, 0, 0};
    for (int b = 0; b < PASS_BLOCKS; ++b) {
        const float4 p = *(const float4*)&part[(size_t)b * EE + 4 * j4];
        s.x += p.x; s.y += p.y; s.z += p.z; s.w += p.w;
    }
    const float4 en = *(const float4*)&edge_norm[4 * j4];
    float4 o{en.x / s.x, en.y / s.y, en.z / s.z, en.w / s.w};
    *(float4*)&col_scale[4 * j4] = o;
}

// ---------------------------------------------------------------------------
// Pack kernel: H[i,:] = scale[i] * (X[i,:] @ W)  written directly in the WMMA
// B-fragment layout for 16x16x32 bf16:
//   fragment id = (i/32)*4 + c/16 ; lane = (c&15) + 16*((i>>3)&1) ;
//   slot = (i&7) + 8*((i>>4)&1) ; each lane's 16 bf16 are contiguous (one 32B load later).
// ---------------------------------------------------------------------------
__global__ __launch_bounds__(256) void pack_h_kernel(const float* __restrict__ X,
                                                     const float* __restrict__ W,
                                                     const float* __restrict__ scale,
                                                     short* __restrict__ Hp) {
    __shared__ float Wl[64 * 64];
    __shared__ float xl[4 * 64];
    const int tid = threadIdx.x;
#pragma unroll
    for (int q = 0; q < 16; ++q) Wl[q * 256 + tid] = W[q * 256 + tid];
    const int i0 = blockIdx.x * 4;
    const int il = tid >> 6, c = tid & 63;
    xl[tid] = X[(size_t)(i0 + il) * CC + c];
    __syncthreads();
    float acc = 0.f;
#pragma unroll
    for (int k = 0; k < 64; ++k) acc += xl[il * 64 + k] * Wl[k * 64 + c];
    const int i = i0 + il;
    const float h = acc * scale[i];
    const int kb = i >> 5, ki = i & 31;
    const int lanei = (c & 15) + ((ki & 8) ? 16 : 0);
    const int slot  = (ki & 7) + ((ki & 16) ? 8 : 0);
    const int nb    = c >> 4;
    Hp[((size_t)((kb * 4 + nb) * 32 + lanei)) * 16 + slot] = f32_to_bf16_rne(h);
}

// ---------------------------------------------------------------------------
// GEMM1: x1 = col_scale .* (B1^T @ H0') + bias0 ; also emits relu(x1).
// Block: 128 thr (4 waves), 64 output edge-rows, K = NN. Double-buffered LDS
// ping-pong (one barrier per K step). Each thread owns a 4(K)x4(e) micro-tile
// of B1: 4 coalesced float4 loads from consecutive K-rows; the transpose is
// emitted as 4x ds_store_b64 of packed bf16 (4 consecutive K per store).
// Tile k feeds ds_load_b128 A fragments into v_wmma_f32_16x16x32_bf16.
// ---------------------------------------------------------------------------
__global__ __launch_bounds__(128) void gemm1_kernel(const float* __restrict__ B1,
                                                    const short* __restrict__ Hp,
                                                    const float* __restrict__ col_scale,
                                                    const float* __restrict__ bias,
                                                    float* __restrict__ x1ws,
                                                    float* __restrict__ out_relu) {
    __shared__ __align__(16) short As[2][GEMM_ROWS * 32];
    const int tid = threadIdx.x;
    const int w = tid >> 5, lane = tid & 31;
    const int m = lane & 15, half = lane >> 4;
    const int e0 = blockIdx.x * GEMM_ROWS;
    v8f acc[4] = {};
    // micro-tile ownership: 4 consecutive K-rows x one float4 of e
    const int n4 = (tid >> 4) * 4;       // 0,4,..,28
    const int e4 = tid & 15;             // float4 index in e (0..15 -> e 0..63)
    float4 stage[4];

    auto load_tile = [&](int k0) {
#pragma unroll
        for (int j = 0; j < 4; ++j)
            stage[j] = *(const float4*)(B1 + (size_t)(k0 + n4 + j) * EE + e0 + 4 * e4);
    };
    auto store_tile = [&](int buf) {     // transpose: As[e*32 + n], 4 n per b64 store
        const float* fp = (const float*)&stage[0];
#pragma unroll
        for (int u = 0; u < 4; ++u) {
            uint2 p;
            p.x = pack_bf16_trunc(fp[0 * 4 + u], fp[1 * 4 + u]);   // n4+0, n4+1
            p.y = pack_bf16_trunc(fp[2 * 4 + u], fp[3 * 4 + u]);   // n4+2, n4+3
            *(uint2*)&As[buf][(4 * e4 + u) * 32 + n4] = p;
        }
    };

    const int KT = NN / 32;
    load_tile(0);
    store_tile(0);
    __syncthreads();
    for (int kt = 0; kt < KT; ++kt) {
        const int cur = kt & 1;
        if (kt + 1 < KT) load_tile(32 * (kt + 1));
        if (kt + 2 < KT)
            __builtin_prefetch(B1 + (size_t)(32 * (kt + 2) + n4) * EE + e0 + 4 * e4, 0, 1);
        const int abase = (w * 16 + m) * 32 + half * 8;
        const s8v alo = *(const s8v*)&As[cur][abase];
        const s8v ahi = *(const s8v*)&As[cur][abase + 16];
        const s16v a16 = __builtin_shufflevector(alo, ahi, 0, 1, 2, 3, 4, 5, 6, 7,
                                                 8, 9, 10, 11, 12, 13, 14, 15);
        const v16bf A = __builtin_bit_cast(v16bf, a16);
#pragma unroll
        for (int nb = 0; nb < 4; ++nb) {
            const s16v b16 = *(const s16v*)(Hp + ((size_t)((kt * 4 + nb) * 32 + lane)) * 16);
            const v16bf Bv = __builtin_bit_cast(v16bf, b16);
            acc[nb] = __builtin_amdgcn_wmma_f32_16x16x32_bf16(
                false, A, false, Bv, (short)0, acc[nb], false, false);
        }
        if (kt + 1 < KT) store_tile((kt + 1) & 1);
        __syncthreads();
    }
#pragma unroll
    for (int nb = 0; nb < 4; ++nb) {
        const int col = nb * 16 + m;
        const float b = bias[col];
#pragma unroll
        for (int r = 0; r < 8; ++r) {
            const int j = e0 + w * 16 + r + 8 * half;   // C layout: M = r + 8*(lane>=16)
            const float val = acc[nb][r] * col_scale[j] + b;
            x1ws[(size_t)j * CC + col]     = val;                     // pre-relu feeds conv2
            out_relu[(size_t)j * CC + col] = val > 0.f ? val : 0.f;   // relu(x_1) output
        }
    }
}

// ---------------------------------------------------------------------------
// GEMM2: x0_out = relu(row_scale .* (B1 @ H1') + bias1). Same structure,
// A tile is B1 row-major: each float4 packs to one ds_store_b64. K = EE.
// ---------------------------------------------------------------------------
__global__ __launch_bounds__(128) void gemm2_kernel(const float* __restrict__ B1,
                                                    const short* __restrict__ Hp,
                                                    const float* __restrict__ row_scale,
                                                    const float* __restrict__ bias,
                                                    float* __restrict__ out0) {
    __shared__ __align__(16) short As[2][GEMM_ROWS * 32];
    const int tid = threadIdx.x;
    const int w = tid >> 5, lane = tid & 31;
    const int m = lane & 15, half = lane >> 4;
    const int i0 = blockIdx.x * GEMM_ROWS;
    v8f acc[4] = {};
    float4 stage[4];

    auto load_tile = [&](int k0) {                 // B1[i0+i, k0+k], i 0..63, k 0..31
#pragma unroll
        for (int q = 0; q < 4; ++q) {
            const int v = q * 128 + tid;
            const int i = v >> 3, k4 = v & 7;
            stage[q] = *(const float4*)(B1 + (size_t)(i0 + i) * EE + k0 + 4 * k4);
        }
    };
    auto store_tile = [&](int buf) {               // row-major: As[i*32 + k]
#pragma unroll
        for (int q = 0; q < 4; ++q) {
            const int v = q * 128 + tid;
            const int i = v >> 3, k4 = v & 7;
            uint2 p;
            p.x = pack_bf16_trunc(stage[q].x, stage[q].y);
            p.y = pack_bf16_trunc(stage[q].z, stage[q].w);
            *(uint2*)&As[buf][i * 32 + 4 * k4] = p;
        }
    };

    const int KT = EE / 32;
    load_tile(0);
    store_tile(0);
    __syncthreads();
    for (int kt = 0; kt < KT; ++kt) {
        const int cur = kt & 1;
        if (kt + 1 < KT) load_tile(32 * (kt + 1));
        if (kt + 2 < KT)
            __builtin_prefetch(B1 + (size_t)(i0 + (tid >> 3)) * EE + 32 * (kt + 2) + 4 * (tid & 7), 0, 1);
        const int abase = (w * 16 + m) * 32 + half * 8;
        const s8v alo = *(const s8v*)&As[cur][abase];
        const s8v ahi = *(const s8v*)&As[cur][abase + 16];
        const s16v a16 = __builtin_shufflevector(alo, ahi, 0, 1, 2, 3, 4, 5, 6, 7,
                                                 8, 9, 10, 11, 12, 13, 14, 15);
        const v16bf A = __builtin_bit_cast(v16bf, a16);
#pragma unroll
        for (int nb = 0; nb < 4; ++nb) {
            const s16v b16 = *(const s16v*)(Hp + ((size_t)((kt * 4 + nb) * 32 + lane)) * 16);
            const v16bf Bv = __builtin_bit_cast(v16bf, b16);
            acc[nb] = __builtin_amdgcn_wmma_f32_16x16x32_bf16(
                false, A, false, Bv, (short)0, acc[nb], false, false);
        }
        if (kt + 1 < KT) store_tile((kt + 1) & 1);
        __syncthreads();
    }
#pragma unroll
    for (int nb = 0; nb < 4; ++nb) {
        const int col = nb * 16 + m;
        const float b = bias[col];
#pragma unroll
        for (int r = 0; r < 8; ++r) {
            const int i = i0 + w * 16 + r + 8 * half;
            const float val = acc[nb][r] * row_scale[i] + b;
            out0[(size_t)i * CC + col] = val > 0.f ? val : 0.f;
        }
    }
}

// ---------------------------------------------------------------------------
extern "C" void kernel_launch(void* const* d_in, const int* in_sizes, int n_in,
                              void* d_out, int out_size, void* d_ws, size_t ws_size,
                              hipStream_t stream) {
    const float* x0  = (const float*)d_in[0];
    const float* B1  = (const float*)d_in[1];
    const float* W0  = (const float*)d_in[2];
    const float* W1  = (const float*)d_in[3];
    const float* b01 = (const float*)d_in[4];
    const float* b10 = (const float*)d_in[5];

    float* out0 = (float*)d_out;            // relu(x_0_out) [NN, CC]
    float* out1 = out0 + (size_t)NN * CC;   // relu(x_1)     [EE, CC]

    // Workspace bump allocator (~24 MB total)
    char* ws = (char*)d_ws;
    float* part      = (float*)ws; ws += (size_t)PASS_BLOCKS * EE * 4;  // 16 MB
    float* edge_norm = (float*)ws; ws += (size_t)EE * 4;
    float* node_norm = (float*)ws; ws += (size_t)NN * 4;
    float* row_scale = (float*)ws; ws += (size_t)NN * 4;
    float* col_scale = (float*)ws; ws += (size_t)EE * 4;
    float* x1ws      = (float*)ws; ws += (size_t)EE * CC * 4;           // 4 MB
    short* H0p       = (short*)ws; ws += (size_t)NN * CC * 2;           // 1 MB
    short* H1p       = (short*)ws; ws += (size_t)EE * CC * 2;           // 2 MB
    (void)ws_size; (void)in_sizes; (void)n_in; (void)out_size;

    const size_t pass_lds = (size_t)(EE + 8) * sizeof(float);           // 64 KB + reduce pad

    // 1) row/col sums (B1 pass #1)
    pass1_kernel<<<PASS_BLOCKS, 256, pass_lds, stream>>>(B1, node_norm, part);
    // 2) edge_norm = edge_sum^-1.5 (deterministic partial reduce, float4)
    reduce_edge_norm_kernel<<<EE / 4 / 256, 256, 0, stream>>>(part, edge_norm);
    // 3) weighted sums (B1 pass #2) -> row_scale + col partials
    pass2_kernel<<<PASS_BLOCKS, 256, pass_lds, stream>>>(B1, edge_norm, node_norm,
                                                         row_scale, part);
    // 4) col_scale = edge_norm / colden
    reduce_col_scale_kernel<<<EE / 4 / 256, 256, 0, stream>>>(part, edge_norm, col_scale);
    // 5) H0' = row_scale .* (x0 @ W0), packed as WMMA B fragments
    pack_h_kernel<<<NN / 4, 256, 0, stream>>>(x0, W0, row_scale, H0p);
    // 6) x1 = col_scale .* (B1^T @ H0') + bias0 (B1 pass #3, WMMA)
    gemm1_kernel<<<EE / GEMM_ROWS, 128, 0, stream>>>(B1, H0p, col_scale, b01, x1ws, out1);
    // 7) H1' = col_scale .* (x1 @ W1), packed
    pack_h_kernel<<<EE / 4, 256, 0, stream>>>(x1ws, W1, col_scale, H1p);
    // 8) x0_out = relu(row_scale .* (B1 @ H1') + bias1) (B1 pass #4, WMMA)
    gemm2_kernel<<<NN / GEMM_ROWS, 128, 0, stream>>>(B1, H1p, row_scale, b10, out0);
}
